// HyperTransformerUNet_6004364280637
// MI455X (gfx1250) — matrix operand order repaired and checked by reading, not compiled, not run on hardware
//
#include <hip/hip_runtime.h>
#include <hip/hip_bf16.h>

typedef __bf16 bf16;
typedef __attribute__((ext_vector_type(16))) __bf16 v16bf;
typedef __attribute__((ext_vector_type(8)))  float  v8f;

#define DIM   192
#define HH    256
#define WW    256
#define BB    2
#define NPIX  (BB*HH*WW)        // 131072
#define NWIN  (BB*64*64)        // 8192
#define KBIG  (16*DIM)          // 3072
#define DHID  (4*DIM)           // 768

// ---------------- workspace layout (bytes) ----------------
// region A: [0, 201326592)  : early = xw bf16 (8192x3072 = 50331648B)
//                             late  = h1 bf16 (131072x768 = 201326592B)
#define OFF_XW   ((size_t)0)
#define OFF_H1   ((size_t)0)
#define OFF_WFT  ((size_t)201326592)                    // wfT bf16 3072x3072 (18874368B)
#define OFF_Y1   (OFF_WFT + 18874368)                   // y1 / xn2 bf16 131072x192 (50331648B)
#define OFF_XR   (OFF_Y1 + 50331648)                    // xr f32 131072x192 (100663296B)
#define OFF_PRJT (OFF_XR + 100663296)                   // projT bf16 192x192
#define OFF_F1T  (OFF_PRJT + 73728)                     // fc1T bf16 768x192
#define OFF_F2T  (OFF_F1T + 294912)                     // fc2T bf16 192x768

// ---------------- helpers ----------------
__device__ __forceinline__ float gelu_exact(float x) {
    return 0.5f * x * (1.0f + erff(x * 0.70710678118654752f));
}

// Cayley-Dickson basis product at n=16: e_i * e_j = sgn * e_k
__device__ __forceinline__ void sed_basis_mul(int i, int j, int& k, float& sgn) {
    int n = 16, kk = 0; float s = 1.0f;
    while (n > 1) {
        int h = n >> 1;
        bool ih = i >= h, jh = j >= h;
        if (!ih && !jh) {
            // c1 = a1*b1 : recurse (i, j)
        } else if (ih && !jh) {
            // c2 = a2 * conj(b1) : sign flips unless j==0
            if (j != 0) s = -s;
            kk += h; i -= h;
        } else if (!ih && jh) {
            // c2 = b2 * a1 : recurse (j-h, i)
            kk += h;
            int ni = j - h, nj = i; i = ni; j = nj;
        } else {
            // c1 = -conj(b2)*a2 : sign flips iff (j-h)==0
            if ((j - h) == 0) s = -s;
            int ni = j - h, nj = i - h; i = ni; j = nj;
        }
        n = h;
    }
    k = kk; sgn = s;
}

// ---------------- kernel: materialize transposed sedenion weight (bf16) ----------------
// wfT[nd][kd] = wf[kd][nd],  kd = i*192+dd (input feat), nd = kb*192+ee (output feat)
__global__ __launch_bounds__(256) void k_build_wfT(const float* __restrict__ w_sed,
                                                   bf16* __restrict__ wfT) {
    int kd = blockIdx.x * 256 + threadIdx.x;   // 0..3071
    int nd = blockIdx.y;                       // 0..3071
    int i  = kd / DIM, dd = kd - i * DIM;
    int kb = nd / DIM, ee = nd - kb * DIM;
    float sign = 0.0f; int jj = 0;
    #pragma unroll 1
    for (int j = 0; j < 16; ++j) {
        int k; float s; sed_basis_mul(i, j, k, s);
        if (k == kb) { jj = j; sign = s; break; }
    }
    float v = sign * w_sed[((size_t)jj * DIM + dd) * DIM + ee];
    wfT[(size_t)nd * KBIG + kd] = (bf16)v;
}

// ---------------- kernel: transpose-convert small weight to bf16 ----------------
// dst[n*K + k] = (bf16) src[k*N + n]
__global__ __launch_bounds__(256) void k_transpose_bf16(const float* __restrict__ src,
                                                        bf16* __restrict__ dst,
                                                        int K, int N) {
    int k = blockIdx.x * 256 + threadIdx.x;
    int n = blockIdx.y;
    if (k < K) dst[(size_t)n * K + k] = (bf16)src[(size_t)k * N + n];
}

// ---------------- kernel: LayerNorm1 (over C) + 4x4 windowing, write bf16 ----------------
// grid (H, B), block 256 (one thread per w)
__global__ __launch_bounds__(256) void k_norm1_window(const float* __restrict__ x,
                                                      const float* __restrict__ g,
                                                      const float* __restrict__ bta,
                                                      bf16* __restrict__ xw) {
    int w = threadIdx.x, h = blockIdx.x, b = blockIdx.y;
    size_t base = ((size_t)b * DIM) * (HH * WW) + (size_t)h * WW + w;
    float s = 0.f, s2 = 0.f;
    #pragma unroll 4
    for (int c = 0; c < DIM; ++c) {
        float v = x[base + (size_t)c * (HH * WW)];
        s += v; s2 += v * v;
    }
    float mean = s * (1.0f / DIM);
    float var  = s2 * (1.0f / DIM) - mean * mean;
    float rstd = rsqrtf(var + 1e-5f);
    size_t row = (size_t)b * 4096 + (size_t)(h >> 2) * 64 + (w >> 2);
    int pbase  = ((h & 3) * 4 + (w & 3)) * DIM;
    bf16* dst = xw + row * KBIG + pbase;
    #pragma unroll 4
    for (int c = 0; c < DIM; ++c) {
        float v = x[base + (size_t)c * (HH * WW)];   // L2-resident re-read
        dst[c] = (bf16)((v - mean) * rstd * g[c] + bta[c]);
    }
}

// ---------------- kernel: LayerNorm2 over contiguous rows (xr f32 -> xn2 bf16) ----------
__global__ __launch_bounds__(256) void k_norm2_rows(const float* __restrict__ xr,
                                                    const float* __restrict__ g,
                                                    const float* __restrict__ bta,
                                                    bf16* __restrict__ xn2) {
    int lane = threadIdx.x & 31;
    int row  = blockIdx.x * 8 + (threadIdx.x >> 5);
    const float* p = xr + (size_t)row * DIM;
    float v[6]; float s = 0.f, s2 = 0.f;
    #pragma unroll
    for (int t = 0; t < 6; ++t) { v[t] = p[lane + 32 * t]; s += v[t]; s2 += v[t] * v[t]; }
    #pragma unroll
    for (int off = 16; off >= 1; off >>= 1) {
        s  += __shfl_xor(s,  off, 32);
        s2 += __shfl_xor(s2, off, 32);
    }
    float mean = s * (1.0f / DIM);
    float var  = s2 * (1.0f / DIM) - mean * mean;
    float rstd = rsqrtf(var + 1e-5f);
    bf16* q = xn2 + (size_t)row * DIM;
    #pragma unroll
    for (int t = 0; t < 6; ++t) {
        int c = lane + 32 * t;
        q[c] = (bf16)((v[t] - mean) * rstd * g[c] + bta[c]);
    }
}

// ---------------- WMMA GEMM: C(MxN) = A(MxK,bf16,row) * Bt(NxK,bf16,row)^T -----------
// each wave computes a (16*TM)x(16*TN) C tile (TMxTN v_wmma_f32_16x16x32_bf16 accs)
// EPI: 0 = gemm1 (window-reverse + exact gelu -> bf16 y1, pixel-major)
//      1 = proj  (+bias +x residual -> f32 xr pixel-major)
//      2 = fc1   (+bias, gelu -> bf16 h1)
//      3 = fc2   (+bias +xr residual -> f32 out in (b,c,h,w))
union FragU { v16bf v; uint4 q[2]; };

template <int EPI, int TM, int TN>
__global__ __launch_bounds__(256) void k_gemm_wmma(const bf16* __restrict__ A,
                                                   const bf16* __restrict__ Bt,
                                                   int M, int N, int K,
                                                   const float* __restrict__ bias,
                                                   const float* __restrict__ resid,
                                                   float* __restrict__ outF,
                                                   bf16* __restrict__ outB) {
    const int waveId = blockIdx.x * 8 + (threadIdx.x >> 5);
    const int tilesN = N / (16 * TN);
    const int tM     = waveId / tilesN;      // supertile row
    const int tN     = waveId - tM * tilesN; // supertile col
    if (tM >= M / (16 * TM)) return;         // wave-uniform: EXEC stays all-ones

    const int lane = threadIdx.x & 31;
    const int hf   = lane >> 4;              // lane half (0/1)
    const int ln   = lane & 15;

    v8f acc[TM][TN] = {};

    const bf16* aBase[TM];
    const bf16* bBase[TN];
    #pragma unroll
    for (int i = 0; i < TM; ++i)
        aBase[i] = A + (size_t)(tM * (16 * TM) + i * 16 + ln) * K + hf * 8;
    #pragma unroll
    for (int j = 0; j < TN; ++j)
        bBase[j] = Bt + (size_t)(tN * (16 * TN) + j * 16 + ln) * K + hf * 8;

    for (int kk = 0; kk < K; kk += 32) {
        FragU a[TM], b[TN];
        #pragma unroll
        for (int i = 0; i < TM; ++i) {
            // A frag: chunk {kk+hf*8 .. +7} and {kk+16+hf*8 .. +7}
            a[i].q[0] = *(const uint4*)(aBase[i] + kk);
            a[i].q[1] = *(const uint4*)(aBase[i] + kk + 16);
        }
        #pragma unroll
        for (int j = 0; j < TN; ++j) {
            // B frag: chunk {kk+hf*8 .. +7} and {kk+16+hf*8 .. +7}
            b[j].q[0] = *(const uint4*)(bBase[j] + kk);
            b[j].q[1] = *(const uint4*)(bBase[j] + kk + 16);
        }
        __builtin_prefetch((const void*)(aBase[0] + kk + 32), 0, 3);
        __builtin_prefetch((const void*)(bBase[0] + kk + 32), 0, 3);

        #pragma unroll
        for (int i = 0; i < TM; ++i)
            #pragma unroll
            for (int j = 0; j < TN; ++j)
                acc[i][j] = __builtin_amdgcn_wmma_f32_16x16x32_bf16(
                    false, a[i].v, false, b[j].v, (short)0, acc[i][j], false, false);
    }

    // ---- epilogue ----
    #pragma unroll
    for (int i = 0; i < TM; ++i) {
        #pragma unroll
        for (int j = 0; j < TN; ++j) {
            #pragma unroll
            for (int r = 0; r < 8; ++r) {
                int row = tM * (16 * TM) + i * 16 + r + hf * 8;  // C layout: m = r + 8*half
                int col = tN * (16 * TN) + j * 16 + ln;          // n = lane&15
                float v = acc[i][j][r];
                if constexpr (EPI == 0) {
                    // row = window idx, col = p*192 + c ; reverse window + exact gelu
                    int b  = row >> 12;
                    int hh = (row >> 6) & 63;
                    int ww = row & 63;
                    int p  = col / DIM;
                    int c  = col - p * DIM;
                    int h  = hh * 4 + (p >> 2);
                    int w  = ww * 4 + (p & 3);
                    size_t pix = ((size_t)b << 16) + (size_t)h * WW + w;
                    outB[pix * DIM + c] = (bf16)gelu_exact(v);
                } else if constexpr (EPI == 1) {
                    // + proj_b + original-x residual -> xr f32 (pixel-major)
                    int b  = row >> 16;
                    int hw = row & 65535;
                    float xv = resid[((size_t)(b * DIM + col)) * (HH * WW) + hw];
                    outF[(size_t)row * DIM + col] = v + bias[col] + xv;
                } else if constexpr (EPI == 2) {
                    outB[(size_t)row * N + col] = (bf16)gelu_exact(v + bias[col]);
                } else {
                    // + fc2_b + xr residual -> final out (b,c,h,w) f32
                    int b  = row >> 16;
                    int hw = row & 65535;
                    float rv = resid[(size_t)row * DIM + col];
                    outF[((size_t)(b * DIM + col)) * (HH * WW) + hw] = v + bias[col] + rv;
                }
            }
        }
    }
}

// ---------------- host ----------------
extern "C" void kernel_launch(void* const* d_in, const int* in_sizes, int n_in,
                              void* d_out, int out_size, void* d_ws, size_t ws_size,
                              hipStream_t stream) {
    (void)in_sizes; (void)n_in; (void)out_size; (void)ws_size;
    const float* x       = (const float*)d_in[0];
    const float* norm1_g = (const float*)d_in[1];
    const float* norm1_b = (const float*)d_in[2];
    const float* w_sed   = (const float*)d_in[3];
    const float* proj_w  = (const float*)d_in[4];
    const float* proj_b  = (const float*)d_in[5];
    const float* norm2_g = (const float*)d_in[6];
    const float* norm2_b = (const float*)d_in[7];
    const float* fc1_w   = (const float*)d_in[8];
    const float* fc1_b   = (const float*)d_in[9];
    const float* fc2_w   = (const float*)d_in[10];
    const float* fc2_b   = (const float*)d_in[11];
    float* out = (float*)d_out;

    char* ws = (char*)d_ws;
    bf16*  xw    = (bf16*)(ws + OFF_XW);
    bf16*  h1    = (bf16*)(ws + OFF_H1);
    bf16*  wfT   = (bf16*)(ws + OFF_WFT);
    bf16*  y1    = (bf16*)(ws + OFF_Y1);
    bf16*  xn2   = (bf16*)(ws + OFF_Y1);   // reuses y1 slot after proj
    float* xr    = (float*)(ws + OFF_XR);
    bf16*  projT = (bf16*)(ws + OFF_PRJT);
    bf16*  f1T   = (bf16*)(ws + OFF_F1T);
    bf16*  f2T   = (bf16*)(ws + OFF_F2T);

    // 1) weights -> bf16 (transposed, N-major)
    k_build_wfT<<<dim3(KBIG / 256, KBIG), 256, 0, stream>>>(w_sed, wfT);
    k_transpose_bf16<<<dim3(1, DIM),  256, 0, stream>>>(proj_w, projT, DIM,  DIM);
    k_transpose_bf16<<<dim3(1, DHID), 256, 0, stream>>>(fc1_w,  f1T,   DIM,  DHID);
    k_transpose_bf16<<<dim3(3, DIM),  256, 0, stream>>>(fc2_w,  f2T,   DHID, DIM);

    // 2) LayerNorm1 + window -> xw bf16 (8192 x 3072)
    k_norm1_window<<<dim3(HH, BB), 256, 0, stream>>>(x, norm1_g, norm1_b, xw);

    // 3) GEMM1: xw @ wf  -> window-reverse + gelu -> y1 bf16 (131072 x 192)
    {
        int tiles = (NWIN / 64) * (KBIG / 64);       // 4x4 supertiles
        k_gemm_wmma<0, 4, 4><<<(tiles + 7) / 8, 256, 0, stream>>>(
            xw, wfT, NWIN, KBIG, KBIG, nullptr, nullptr, nullptr, y1);
    }
    // 4) proj: y1 @ proj_w + proj_b + x -> xr f32 (131072 x 192)
    {
        int tiles = (NPIX / 64) * (DIM / 32);        // 4x2 supertiles
        k_gemm_wmma<1, 4, 2><<<(tiles + 7) / 8, 256, 0, stream>>>(
            y1, projT, NPIX, DIM, DIM, proj_b, x, xr, nullptr);
    }
    // 5) LayerNorm2 -> xn2 bf16
    k_norm2_rows<<<NPIX / 8, 256, 0, stream>>>(xr, norm2_g, norm2_b, xn2);

    // 6) fc1 + gelu -> h1 bf16 (131072 x 768)   [reuses xw region]
    {
        int tiles = (NPIX / 64) * (DHID / 64);       // 4x4 supertiles
        k_gemm_wmma<2, 4, 4><<<(tiles + 7) / 8, 256, 0, stream>>>(
            xn2, f1T, NPIX, DHID, DIM, fc1_b, nullptr, nullptr, h1);
    }
    // 7) fc2 + bias + xr residual -> d_out (b,c,h,w) f32
    {
        int tiles = (NPIX / 64) * (DIM / 32);        // 4x2 supertiles
        k_gemm_wmma<3, 4, 2><<<(tiles + 7) / 8, 256, 0, stream>>>(
            h1, f2T, NPIX, DIM, DHID, fc2_b, xr, out, nullptr);
    }
}
